// SemiConnectedConv_37142877176169
// MI455X (gfx1250) — matrix operand order repaired and checked by reading, not compiled, not run on hardware
//
#include <hip/hip_runtime.h>

typedef __attribute__((ext_vector_type(16))) _Float16      v16h;
typedef __attribute__((ext_vector_type(8)))  float         v8f;
typedef __attribute__((ext_vector_type(4)))  unsigned int  v4u;
typedef __attribute__((ext_vector_type(4)))  float         v4f;

#define C_IN   16
#define N_BR   32
#define KF     5
#define HH     224
#define WW     224
#define BB     16
#define NGRP   (C_IN * KF)               // 80 (c,kh) groups
#define TAP    8                         // taps padded 5 -> 8
#define WPAD_ELEMS (N_BR * NGRP * TAP)   // 20480 halves = 40 KB
#define ROW_DW 256                       // dwords per G row: 128 even + 128 odd
#define EVEN_DW 128
#define XS_DW  (NGRP * ROW_DW)           // 20480 dwords = 80 KB
#define NCHUNK 20                        // K = 640 = 20 x 32
#define THREADS 256                      // 8 waves
#define WTILES 14

// ---------------------------------------------------------------------------
// Densify grouped-conv weights into Wpad[n][G=c*5+kh][tap] (f16); zero where
// channel c is not selected by branch n or tap >= 5.
// ---------------------------------------------------------------------------
__global__ __launch_bounds__(256) void init_wpad_kernel(
    const float* __restrict__ Wc, _Float16* __restrict__ wpad) {
  int idx = blockIdx.x * 256 + threadIdx.x;
  if (idx >= WPAD_ELEMS) return;
  int kw = idx & 7;
  int G  = (idx >> 3) % NGRP;
  int n  = idx / (NGRP * TAP);
  int c  = G / KF;
  int kh = G % KF;
  float v = 0.0f;
  if (kw < KF && (((c + n) & 15) < 8)) {
    int j = 0;                              // rank of c among selected channels
    for (int cp = 0; cp < c; ++cp) j += (((cp + n) & 15) < 8) ? 1 : 0;
    v = Wc[((n * 8 + j) * KF + kh) * KF + kw];
  }
  wpad[idx] = (_Float16)v;
}

// ---------------------------------------------------------------------------
// Block = one (b,h) output row. 8 waves; waves 0..5 own 2 w-tiles, waves 6..7
// own 1 tile each. Dual-phase row buffer: R[t] = x(w = t-2); even-region dword
// p = halves (R[2p],R[2p+1]); odd-region dword p = halves (R[2p+1],R[2p+2]).
// A lane's 8-half window is then 4 contiguous dwords at a per-lane-constant
// phase offset -> pure ds loads, zero VALU in the K loop.
// ---------------------------------------------------------------------------
__global__ __launch_bounds__(THREADS) void semiconv_wmma_kernel(
    const float* __restrict__ x, const float* __restrict__ bc,
    const _Float16* __restrict__ wpad_g, float* __restrict__ out) {
  __shared__ _Float16     wp[WPAD_ELEMS];   // 40 KB densified weights
  __shared__ unsigned int xs[XS_DW];        // 80 KB dual-phase row buffer

  const int tid  = threadIdx.x;
  const int wv   = tid >> 5;
  const int lane = tid & 31;
  const int n    = lane & 15;   // N within tile / pixel row M
  const int hi   = lane >> 4;   // lane half

  const int h = blockIdx.x % HH;
  const int b = blockIdx.x / HH;

  // ---- stage densified weights block-wide (2560 x 16B) ----
  {
    const v4u* __restrict__ src = (const v4u*)wpad_g;
    v4u* dst = (v4u*)wp;
    for (int i = tid; i < WPAD_ELEMS / 8; i += THREADS) dst[i] = src[i];
  }

  // ---- pass 1: raw rows (f32 -> f16) into even region; t == tid ----
  {
    _Float16* xsh = (_Float16*)xs;
    const int  t   = tid;                       // R index; w = t - 2
    const int  w   = t - 2;
    const bool wok = (unsigned)w < (unsigned)WW;
    const int  wc  = w < 0 ? 0 : (w >= WW ? WW - 1 : w);   // clamped (valid addr)
    const float* xbase = x + (size_t)b * C_IN * HH * WW + wc;
#pragma unroll
    for (int kh = 0; kh < KF; ++kh) {
      const int hs = h + kh - 2;                // wave-uniform
      if ((unsigned)hs < (unsigned)HH) {
        const float* xp = xbase + (size_t)hs * WW;
#pragma unroll
        for (int c = 0; c < C_IN; ++c) {
          float v = xp[(size_t)c * HH * WW];
          v = wok ? v : 0.0f;
          xsh[(c * KF + kh) * (2 * ROW_DW) + t] = (_Float16)v;
        }
      } else {
#pragma unroll
        for (int c = 0; c < C_IN; ++c)
          xsh[(c * KF + kh) * (2 * ROW_DW) + t] = (_Float16)0.0f;
      }
    }
  }
  __syncthreads();

  // ---- pass 2: build odd-phase copy (LDS -> LDS), pure bit-op indexing ----
  {
    const int p  = tid & 127;
    const int g0 = tid >> 7;          // 0 or 1
#pragma unroll 4
    for (int k = 0; k < NGRP / 2; ++k) {
      const int G = g0 + 2 * k;
      unsigned int d0 = xs[G * ROW_DW + p];
      unsigned int d1 = xs[G * ROW_DW + p + 1];  // p==127 reads odd region: discarded
      unsigned int v  = (d0 >> 16) | (d1 << 16);
      xs[G * ROW_DW + EVEN_DW + p] = (p < 127) ? v : 0u;
    }
  }
  __syncthreads();

  // ---- main K loop: wave owns w-tiles ta, tb; B fragments shared ----
  const int ta = (wv < 6) ? 2 * wv : (6 + wv);      // wv6 -> 12, wv7 -> 13
  const int tb = (wv < 6) ? 2 * wv + 1 : (6 + wv);  // duplicate pair for wv>=6
  const int laneoff = (n & 1) * EVEN_DW + (n >> 1); // phase + base dword

  v8f accA0 = {}, accA1 = {}, accB0 = {}, accB1 = {};

#pragma unroll
  for (int chunk = 0; chunk < NCHUNK; ++chunk) {
    const int Gc = chunk * 4;

    // B fragments: lane(n,hi) holds Wpad[N][Gc+2hi][0..15] (16 contiguous halves)
    union { v4u u[2]; v16h hv; } bu0, bu1;
    const v4u* bp0 = (const v4u*)(wp + ((n)      * NGRP + Gc + 2 * hi) * TAP);
    const v4u* bp1 = (const v4u*)(wp + ((16 + n) * NGRP + Gc + 2 * hi) * TAP);
    bu0.u[0] = bp0[0]; bu0.u[1] = bp0[1];
    bu1.u[0] = bp1[0]; bu1.u[1] = bp1[1];

    // A rows for this lane: groups Gc+2hi (j<8) and Gc+2hi+1 (j>=8)
    const unsigned int* r0 = xs + (Gc + 2 * hi)     * ROW_DW + laneoff;
    const unsigned int* r1 = xs + (Gc + 2 * hi + 1) * ROW_DW + laneoff;

    union { unsigned int u[8]; v16h hv; } aa, ab;
    {
      const int p = ta * 8;
      aa.u[0] = r0[p + 0]; aa.u[1] = r0[p + 1]; aa.u[2] = r0[p + 2]; aa.u[3] = r0[p + 3];
      aa.u[4] = r1[p + 0]; aa.u[5] = r1[p + 1]; aa.u[6] = r1[p + 2]; aa.u[7] = r1[p + 3];
    }
    {
      const int p = tb * 8;
      ab.u[0] = r0[p + 0]; ab.u[1] = r0[p + 1]; ab.u[2] = r0[p + 2]; ab.u[3] = r0[p + 3];
      ab.u[4] = r1[p + 0]; ab.u[5] = r1[p + 1]; ab.u[6] = r1[p + 2]; ab.u[7] = r1[p + 3];
    }

    accA0 = __builtin_amdgcn_wmma_f32_16x16x32_f16(
        false, aa.hv, false, bu0.hv, (short)0, accA0, false, false);
    accA1 = __builtin_amdgcn_wmma_f32_16x16x32_f16(
        false, aa.hv, false, bu1.hv, (short)0, accA1, false, false);
    accB0 = __builtin_amdgcn_wmma_f32_16x16x32_f16(
        false, ab.hv, false, bu0.hv, (short)0, accB0, false, false);
    accB1 = __builtin_amdgcn_wmma_f32_16x16x32_f16(
        false, ab.hv, false, bu1.hv, (short)0, accB1, false, false);
  }

  // ---- epilogue: bias + sigmoid, 16B vector stores ----
  // D layout: VGPR r -> (M = r + 8*hi, N = n); consecutive r -> consecutive w.
  const float bias0 = bc[n];
  const float bias1 = bc[16 + n];
  const int M0 = 8 * hi;

#pragma unroll
  for (int tsel = 0; tsel < 2; ++tsel) {
    const int   tile = tsel ? tb : ta;
    const v8f   a0   = tsel ? accB0 : accA0;
    const v8f   a1   = tsel ? accB1 : accA1;
    float* o0 = out + ((size_t)(b * N_BR + n)      * HH + h) * WW + tile * 16 + M0;
    float* o1 = out + ((size_t)(b * N_BR + 16 + n) * HH + h) * WW + tile * 16 + M0;
    v4f s;
#pragma unroll
    for (int half = 0; half < 2; ++half) {
#pragma unroll
      for (int r = 0; r < 4; ++r) {
        float z = a0[half * 4 + r] + bias0;
        s[r] = 1.0f / (1.0f + __expf(-z));
      }
      *(v4f*)(o0 + half * 4) = s;
#pragma unroll
      for (int r = 0; r < 4; ++r) {
        float z = a1[half * 4 + r] + bias1;
        s[r] = 1.0f / (1.0f + __expf(-z));
      }
      *(v4f*)(o1 + half * 4) = s;
    }
  }
}

extern "C" void kernel_launch(void* const* d_in, const int* in_sizes, int n_in,
                              void* d_out, int out_size, void* d_ws, size_t ws_size,
                              hipStream_t stream) {
  const float* x    = (const float*)d_in[0];   // (16,16,224,224) f32
  const float* Wc   = (const float*)d_in[1];   // (32,8,5,5) f32
  const float* bc   = (const float*)d_in[2];   // (32,) f32
  float*       out  = (float*)d_out;           // (16,32,224,224) f32
  _Float16*    wpad = (_Float16*)d_ws;         // 40 KB densified f16 weights

  init_wpad_kernel<<<(WPAD_ELEMS + 255) / 256, 256, 0, stream>>>(Wc, wpad);

  // one block per (b, h) output row
  semiconv_wmma_kernel<<<BB * HH, THREADS, 0, stream>>>(x, bc, wpad, out);
}
